// Foveater2_16277926052664
// MI455X (gfx1250) — matrix operand order repaired and checked by reading, not compiled, not run on hardware
//
#include <hip/hip_runtime.h>
#include <hip/hip_bf16.h>

// ---------------------------------------------------------------------------
// Problem constants (match reference)
// ---------------------------------------------------------------------------
#define BATCH    64
#define CHAN     3
#define IMG      512
#define RESIZE   64
#define EMB      1024
#define KTOT     49152            // 4 * 3 * 64 * 64
#define KSPLIT   32
#define KCHUNK   (KTOT / KSPLIT)  // 1536

typedef __attribute__((ext_vector_type(16))) __bf16        v16bf;
typedef __attribute__((ext_vector_type(8)))  float         v8f;
typedef __attribute__((ext_vector_type(4)))  unsigned int  u32x4;

__device__ __forceinline__ unsigned f2bfbits(float f) {
    // round-to-nearest-even fp32 -> bf16 bits
    unsigned u = __float_as_uint(f);
    return (u + 0x7FFFu + ((u >> 16) & 1u)) >> 16;
}

// ---------------------------------------------------------------------------
// Kernel 1: fused foveated crop + antialiased separable resize -> bf16 feats
// Triangle kernel support <= 2*kw+1 (<=13 taps for P=384): iterate only the
// support, normalized weights computed on the fly (resize mats are not inputs).
// Grid: (B, C, 4 levels), block 256. Output: feats_bf16[B][KTOT] row-major.
// ---------------------------------------------------------------------------
__device__ __forceinline__ void mk_weights(int o, int P, float scale, float kw,
                                           float inv_kw, float* w, int& jlo,
                                           int& cnt) {
    float center = o * scale;
    int lo = (int)ceilf(center - kw);
    if (lo < 0) lo = 0;
    int hi = (int)floorf(center + kw);
    if (hi > P - 1) hi = P - 1;
    cnt = hi - lo + 1;
    if (cnt > 16) cnt = 16;
    jlo = lo;
    float s = 0.0f;
    for (int i = 0; i < cnt; ++i) {
        float ww = 1.0f - fabsf((float)(lo + i) - center) * inv_kw;
        ww = fmaxf(ww, 0.0f);
        w[i] = ww;
        s += ww;
    }
    float inv = 1.0f / s;
    for (int i = 0; i < cnt; ++i) w[i] *= inv;
}

__global__ __launch_bounds__(256) void foveate_kernel(
    const float* __restrict__ img, const int* __restrict__ px,
    const int* __restrict__ py, unsigned short* __restrict__ featsBf) {
    const int b = blockIdx.x;
    const int c = blockIdx.y;
    const int lvl = blockIdx.z;
    const int Ps[4]    = {64, 128, 256, 384};
    const int Poff[4]  = {0, 12288, 24576, 36864};
    const int P = Ps[lvl];
    const float scale = (float)(P - 1) / 63.0f;
    const float kw = fmaxf(scale, 1.0f);
    const float inv_kw = 1.0f / kw;
    const int cx = px[b];
    const int cy = py[b];
    const float* imc = img + ((size_t)(b * CHAN + c)) * IMG * IMG;
    unsigned short* outp = featsBf + (size_t)b * KTOT + Poff[lvl] + c * 4096;

    for (int e = threadIdx.x; e < RESIZE * RESIZE; e += 256) {
        const int o = e >> 6;     // output row
        const int r = e & 63;     // output col
        float wy[16], wx[16];
        int plo, pcnt, qlo, qcnt;
        mk_weights(o, P, scale, kw, inv_kw, wy, plo, pcnt);
        mk_weights(r, P, scale, kw, inv_kw, wx, qlo, qcnt);
        float acc = 0.0f;
        for (int i = 0; i < pcnt; ++i) {
            const int iy = cy - P / 2 + plo + i;
            if ((unsigned)iy >= (unsigned)IMG) continue;   // PAD_VALUE = 0
            const float* rowp = imc + (size_t)iy * IMG;
            float racc = 0.0f;
            for (int j = 0; j < qcnt; ++j) {
                const int ix = cx - P / 2 + qlo + j;
                const float v = ((unsigned)ix < (unsigned)IMG) ? rowp[ix] : 0.0f;
                racc = fmaf(wx[j], v, racc);
            }
            acc = fmaf(wy[i], racc, acc);
        }
        outp[o * 64 + r] = (unsigned short)f2bfbits(acc);
    }
}

// ---------------------------------------------------------------------------
// Kernel 2: split-K WMMA GEMM, one wave per (ntile, ktile).
// Wave computes full M=64 (4 accumulator tiles) x one 16-wide N slice, so W
// is streamed from HBM exactly once (201 MB -> the 8.6us roofline floor).
// A (feats, bf16): double-buffered 64x32 tile staged into LDS with
//   global_load_async_to_lds_b128 (ASYNCcnt); next tile's copies are issued
//   before computing on the current one, so s_wait_asynccnt 0x8 only waits
//   for the current tile (async loads complete in order). Fragment reads are
//   opaque asm ds_load_b128 with the s_wait_dscnt folded into the same asm
//   block so the compiler can neither delete them nor hoist the WMMAs.
// B (W, fp32): 16 column loads per lane, RNE-converted to bf16 in VALU
//   (co-executes with the XDL WMMAs), issued before the async wait.
// ---------------------------------------------------------------------------
union AFrag { u32x4 q[2]; v16bf v; };
union BFrag { unsigned u[8]; v16bf v; };

__global__ __launch_bounds__(32) void gemm_kernel(
    const float* __restrict__ Wm, const unsigned short* __restrict__ featsBf,
    float* __restrict__ partial) {
    // Double-buffered A tile: 2 x (64 rows x 32 k x bf16) = 8 KB.
    // Only LDS object in this kernel -> wave LDS offset 0; async-to-LDS dst
    // offsets and ds_load addresses below are raw offsets from the LDS base.
    __shared__ __align__(16) unsigned char smem[8192];

    // Force LDS allocation: smem is otherwise only touched via opaque asm.
    asm volatile("" :: "v"((unsigned)(size_t)&smem[0]));

    const int lane  = threadIdx.x;          // 0..31
    const int ntile = blockIdx.x;           // 0..63  (N/16)
    const int kt    = blockIdx.y;           // 0..KSPLIT-1
    const int n0    = ntile * 16;
    const int kstart = kt * KCHUNK;
    const bool hiHalf = lane >= 16;
    const int lrow = lane & 15;

    // Fragment addressing (CDNA5 ISA):
    // A (16x32 bf16): lane<16 -> K {0..7,16..23}; lane>=16 -> K {8..15,24..31}
    const int aoff  = hiHalf ? 8 : 0;
    // B (32x16 bf16): lane<16 -> K {0..7,16..23}; lane>=16 -> K {8..15,24..31}; col = lane&15
    const int kboff = hiHalf ? 8 : 0;
    const int ncol  = n0 + lrow;

    // Per-lane async-copy addressing: chunk ch = it*32+lane (16B chunks of a
    // row-major 64x32 bf16 tile): row = it*8 + (lane>>2), k-off = (lane&3)*8.
    const int crow = lane >> 2;
    const int cko  = (lane & 3) * 8;

    v8f acc0 = {}, acc1 = {}, acc2 = {}, acc3 = {};

    auto issue_tile = [&](int kb, int bb) {
#pragma unroll
        for (int it = 0; it < 8; ++it) {
            const unsigned short* src =
                featsBf + (size_t)(it * 8 + crow) * KTOT + kb + cko;
            const unsigned dst = (unsigned)(bb * 4096 + (it * 32 + lane) * 16);
            asm volatile("global_load_async_to_lds_b128 %0, %1, off"
                         :: "v"(dst),
                            "v"((unsigned long long)(uintptr_t)src)
                         : "memory");
        }
    };

    issue_tile(kstart, 0);                      // prologue: tile 0 -> buf 0

    int step = 0;
    for (int kb = kstart; kb < kstart + KCHUNK; kb += 32, ++step) {
        const int bb = step & 1;
        int nkb = kb + 32;
        if (nkb >= kstart + KCHUNK) nkb = kstart;   // deterministic dummy wrap
        issue_tile(nkb, bb ^ 1);                    // prefetch next tile

        // prefetch W two k-steps ahead (global_prefetch_b8)
        __builtin_prefetch(Wm + (size_t)(kb + kboff + 64) * EMB + ncol, 0, 1);

        // ---- B fragment: issue fp32 column loads before the async wait ----
        const float* wp = Wm + (size_t)(kb + kboff) * EMB + ncol;
        float col[16];
#pragma unroll
        for (int v = 0; v < 16; ++v) col[v] = wp[(size_t)(v + (v & 8)) * EMB];
        BFrag bf;
#pragma unroll
        for (int v = 0; v < 8; ++v)
            bf.u[v] = f2bfbits(col[2 * v]) | (f2bfbits(col[2 * v + 1]) << 16);

        // Wait until only the 8 just-issued (next-tile) copies are pending:
        // current tile's 8 copies have landed in LDS.
        asm volatile("s_wait_asynccnt 0x8" ::: "memory");

        // ---- A fragments from LDS: opaque ds_load_b128 + folded dscnt wait
        const unsigned abase = (unsigned)(bb * 4096 + lrow * 64 + aoff * 2);
        u32x4 q0, q1, q2, q3, q4, q5, q6, q7;
        asm volatile(
            "ds_load_b128 %0, %8\n\t"
            "ds_load_b128 %1, %8 offset:32\n\t"
            "ds_load_b128 %2, %8 offset:1024\n\t"
            "ds_load_b128 %3, %8 offset:1056\n\t"
            "ds_load_b128 %4, %8 offset:2048\n\t"
            "ds_load_b128 %5, %8 offset:2080\n\t"
            "ds_load_b128 %6, %8 offset:3072\n\t"
            "ds_load_b128 %7, %8 offset:3104\n\t"
            "s_wait_dscnt 0x0"
            : "=&v"(q0), "=&v"(q1), "=&v"(q2), "=&v"(q3),
              "=&v"(q4), "=&v"(q5), "=&v"(q6), "=&v"(q7)
            : "v"(abase)
            : "memory");

        AFrag a0, a1, a2, a3;
        a0.q[0] = q0; a0.q[1] = q1;
        a1.q[0] = q2; a1.q[1] = q3;
        a2.q[0] = q4; a2.q[1] = q5;
        a3.q[0] = q6; a3.q[1] = q7;

        acc0 = __builtin_amdgcn_wmma_f32_16x16x32_bf16(
            false, a0.v, false, bf.v, (short)0, acc0, false, false);
        acc1 = __builtin_amdgcn_wmma_f32_16x16x32_bf16(
            false, a1.v, false, bf.v, (short)0, acc1, false, false);
        acc2 = __builtin_amdgcn_wmma_f32_16x16x32_bf16(
            false, a2.v, false, bf.v, (short)0, acc2, false, false);
        acc3 = __builtin_amdgcn_wmma_f32_16x16x32_bf16(
            false, a3.v, false, bf.v, (short)0, acc3, false, false);
    }

    // drain the dummy wrap prefetch before wave end
    asm volatile("s_wait_asynccnt 0x0" ::: "memory");

    // C/D layout: VGPR i, lane<16 -> (M=i, N=lane); lane>=16 -> (M=i+8, N=lane-16)
    const int rbase = hiHalf ? 8 : 0;
    float* pp = partial + ((size_t)kt * BATCH) * EMB + n0 + lrow;
#pragma unroll
    for (int i = 0; i < 8; ++i) {
        pp[(size_t)(0 * 16 + i + rbase) * EMB] = acc0[i];
        pp[(size_t)(1 * 16 + i + rbase) * EMB] = acc1[i];
        pp[(size_t)(2 * 16 + i + rbase) * EMB] = acc2[i];
        pp[(size_t)(3 * 16 + i + rbase) * EMB] = acc3[i];
    }
}

// ---------------------------------------------------------------------------
// Kernel 3: reduce split-K partials + bias + pos-emb gather -> out[B][EMB]
// ---------------------------------------------------------------------------
__global__ __launch_bounds__(256) void reduce_kernel(
    const float* __restrict__ partial, const float* __restrict__ bias,
    const float* __restrict__ pos, const int* __restrict__ px,
    const int* __restrict__ py, float* __restrict__ out) {
    const int idx = blockIdx.x * 256 + threadIdx.x;   // 0..65535
    const int b = idx >> 10;
    const int n = idx & (EMB - 1);
    float s = bias[n];
#pragma unroll
    for (int kt = 0; kt < KSPLIT; ++kt)
        s += partial[((size_t)kt * BATCH + b) * EMB + n];
    s += pos[((size_t)py[b] * IMG + px[b]) * EMB + n];
    out[idx] = s;
}

// ---------------------------------------------------------------------------
// Launch
// ---------------------------------------------------------------------------
extern "C" void kernel_launch(void* const* d_in, const int* in_sizes, int n_in,
                              void* d_out, int out_size, void* d_ws,
                              size_t ws_size, hipStream_t stream) {
    (void)in_sizes; (void)n_in; (void)out_size; (void)ws_size;
    const float* img  = (const float*)d_in[0];   // [64,3,512,512]
    const int*   px   = (const int*)d_in[1];     // [64]
    const int*   py   = (const int*)d_in[2];     // [64]
    const float* pos  = (const float*)d_in[3];   // [512,512,1024]
    const float* Wm   = (const float*)d_in[4];   // [49152,1024]
    const float* bias = (const float*)d_in[5];   // [1024]
    float* out = (float*)d_out;                  // [64,1024]

    // Workspace layout:
    //   featsBf : 64*49152 bf16    = 6,291,456 B
    //   partial : 32*64*1024 f32   = 8,388,608 B
    unsigned short* featsBf = (unsigned short*)d_ws;
    float* partial = (float*)((char*)d_ws + (size_t)BATCH * KTOT * 2);

    foveate_kernel<<<dim3(BATCH, CHAN, 4), 256, 0, stream>>>(img, px, py,
                                                             featsBf);
    gemm_kernel<<<dim3(EMB / 16, KSPLIT), 32, 0, stream>>>(Wm, featsBf,
                                                           partial);
    reduce_kernel<<<dim3(BATCH * EMB / 256), 256, 0, stream>>>(
        partial, bias, pos, px, py, out);
}
